// MultiHeadAttention_5239860101377
// MI455X (gfx1250) — matrix operand order repaired and checked by reading, not compiled, not run on hardware
//
#include <hip/hip_runtime.h>
#include <hip/hip_bf16.h>
#include <math.h>

// ---------------- types for gfx1250 WMMA ----------------
typedef __bf16 bf16;
typedef bf16         v16bf __attribute__((ext_vector_type(16)));
typedef bf16         v4bf  __attribute__((ext_vector_type(4)));
typedef float        v8f   __attribute__((ext_vector_type(8)));
typedef unsigned int v4u   __attribute__((ext_vector_type(4)));

union FragBF { v16bf v; v4u q[2]; };

__device__ __forceinline__ v4bf cvt4(float4 f) {
  v4bf r;
  r[0] = (bf16)f.x; r[1] = (bf16)f.y; r[2] = (bf16)f.z; r[3] = (bf16)f.w;
  return r;   // v_cvt_pk_bf16_f32 pairs on gfx1250
}

// ---------------- problem constants ----------------
static constexpr int BATCH = 4;
static constexpr int SEQ   = 2048;
static constexpr int HEADS = 16;
static constexpr int DK    = 64;      // == DV
static constexpr int DM    = 1024;
static constexpr float LN_EPS = 1e-5f;

// =====================================================================
// Generic GEMM: C[M,N] = A[M,K] @ W[K,N] + bias   (fp32 in, bf16 WMMA)
// 256 threads (8 waves). Tile M=128, N=64, Kstep=32, double-buffered LDS.
// =====================================================================
__global__ __launch_bounds__(256) void wmma_gemm_bias(
    const float* __restrict__ A, const float* __restrict__ W,
    const float* __restrict__ bias, float* __restrict__ C,
    int M, int N, int K)
{
  __shared__ alignas(16) bf16 As[2][128][40];
  __shared__ alignas(16) bf16 Bs[2][64][40];

  const int tid  = threadIdx.x;
  const int lane = tid & 31;
  const int wave = tid >> 5;
  const int M0 = blockIdx.y * 128;
  const int N0 = blockIdx.x * 64;

  v8f acc[4] = {};

  const int rA  = wave * 16 + (lane & 15);
  const int kbA = (lane >> 4) * 8;     // A frag: K in [kb,kb+8) U [kb+16,kb+24)
  const int nB  = lane & 15;
  const int khB = (lane >> 4) * 8;     // B frag: K in [kb,kb+8) U [kb+16,kb+24)

  float4 fa[4], fb[2];

  auto load_regs = [&](int kk) {
    #pragma unroll
    for (int i = 0; i < 4; ++i) {
      int idx = i * 256 + tid;                 // 0..1023 float4 slots
      int r = idx >> 3, c4 = (idx & 7) * 4;
      fa[i] = *reinterpret_cast<const float4*>(
          &A[(size_t)(M0 + r) * K + kk + c4]);
    }
    #pragma unroll
    for (int i = 0; i < 2; ++i) {
      int idx = i * 256 + tid;                 // 0..511 float4 slots
      int kr = idx >> 4, nc = (idx & 15) * 4;
      fb[i] = *reinterpret_cast<const float4*>(
          &W[(size_t)(kk + kr) * N + N0 + nc]);
    }
  };
  auto store_lds = [&](int buf) {
    #pragma unroll
    for (int i = 0; i < 4; ++i) {
      int idx = i * 256 + tid;
      int r = idx >> 3, c4 = (idx & 7) * 4;
      *reinterpret_cast<v4bf*>(&As[buf][r][c4]) = cvt4(fa[i]);
    }
    #pragma unroll
    for (int i = 0; i < 2; ++i) {
      int idx = i * 256 + tid;
      int kr = idx >> 4, nc = (idx & 15) * 4;
      Bs[buf][nc + 0][kr] = (bf16)fb[i].x;
      Bs[buf][nc + 1][kr] = (bf16)fb[i].y;
      Bs[buf][nc + 2][kr] = (bf16)fb[i].z;
      Bs[buf][nc + 3][kr] = (bf16)fb[i].w;
    }
  };

  const int nsteps = K >> 5;
  load_regs(0);
  store_lds(0);
  __syncthreads();

  for (int s = 0; s < nsteps; ++s) {
    const int buf = s & 1;
    if (s + 1 < nsteps) load_regs((s + 1) << 5);   // overlap with WMMA below
    if (s + 2 < nsteps)                            // global_prefetch_b8 (L2 warm)
      __builtin_prefetch(&A[(size_t)(M0 + rA) * K + ((s + 2) << 5)], 0, 1);

    FragBF a;
    a.q[0] = *reinterpret_cast<const v4u*>(&As[buf][rA][kbA]);
    a.q[1] = *reinterpret_cast<const v4u*>(&As[buf][rA][kbA + 16]);
    #pragma unroll
    for (int t = 0; t < 4; ++t) {
      FragBF b;
      b.q[0] = *reinterpret_cast<const v4u*>(&Bs[buf][t * 16 + nB][khB]);
      b.q[1] = *reinterpret_cast<const v4u*>(&Bs[buf][t * 16 + nB][khB + 16]);
      acc[t] = __builtin_amdgcn_wmma_f32_16x16x32_bf16(
          false, a.v, false, b.v, (short)0, acc[t], false, false);
    }

    if (s + 1 < nsteps) store_lds(buf ^ 1);        // other buffer: no conflict
    __syncthreads();
  }

  const int mBase = wave * 16 + (lane >> 4) * 8;   // C/D layout
  const int nLane = lane & 15;
  #pragma unroll
  for (int t = 0; t < 4; ++t) {
    int col = N0 + t * 16 + nLane;
    float bv = bias ? bias[col] : 0.0f;
    #pragma unroll
    for (int r = 0; r < 8; ++r)
      C[(size_t)(M0 + mBase + r) * N + col] = acc[t][r] + bv;
  }
}

// =====================================================================
// Scores: logits[b,h,sq,sk] = (q . k) / sqrt(DK), masked.
// Only 2 k-steps; batch loads into registers, then convert+store.
// =====================================================================
__global__ __launch_bounds__(256) void wmma_scores(
    const float* __restrict__ q, const float* __restrict__ k,
    const int* __restrict__ mask, float* __restrict__ attn)
{
  __shared__ alignas(16) bf16 As[128][40];
  __shared__ alignas(16) bf16 Bs[64][40];

  const int tid  = threadIdx.x;
  const int lane = tid & 31;
  const int wave = tid >> 5;
  const int sq0 = blockIdx.y * 128;
  const int sk0 = blockIdx.x * 64;
  const int bh  = blockIdx.z;
  const int b = bh / HEADS, h = bh % HEADS;

  const float* qb = q + (size_t)b * SEQ * DM + h * DK;
  const float* kb = k + (size_t)b * SEQ * DM + h * DK;

  v8f acc[4] = {};
  const int rA  = wave * 16 + (lane & 15);
  const int kbA = (lane >> 4) * 8;
  const int nB  = lane & 15;
  const int khB = (lane >> 4) * 8;

  #pragma unroll
  for (int kk = 0; kk < DK; kk += 32) {
    float4 fa[4], fb[2];
    #pragma unroll
    for (int i = 0; i < 4; ++i) {
      int idx = i * 256 + tid;
      int r = idx >> 3, c4 = (idx & 7) * 4;
      fa[i] = *reinterpret_cast<const float4*>(
          &qb[(size_t)(sq0 + r) * DM + kk + c4]);
    }
    #pragma unroll
    for (int i = 0; i < 2; ++i) {
      int idx = i * 256 + tid;                 // memory [sk][d] == [n][k]
      int nc = idx >> 3, c4 = (idx & 7) * 4;
      fb[i] = *reinterpret_cast<const float4*>(
          &kb[(size_t)(sk0 + nc) * DM + kk + c4]);
    }
    #pragma unroll
    for (int i = 0; i < 4; ++i) {
      int idx = i * 256 + tid;
      int r = idx >> 3, c4 = (idx & 7) * 4;
      *reinterpret_cast<v4bf*>(&As[r][c4]) = cvt4(fa[i]);
    }
    #pragma unroll
    for (int i = 0; i < 2; ++i) {
      int idx = i * 256 + tid;
      int nc = idx >> 3, c4 = (idx & 7) * 4;
      *reinterpret_cast<v4bf*>(&Bs[nc][c4]) = cvt4(fb[i]);
    }
    __syncthreads();

    FragBF a;
    a.q[0] = *reinterpret_cast<const v4u*>(&As[rA][kbA]);
    a.q[1] = *reinterpret_cast<const v4u*>(&As[rA][kbA + 16]);
    #pragma unroll
    for (int t = 0; t < 4; ++t) {
      FragBF bfr;
      bfr.q[0] = *reinterpret_cast<const v4u*>(&Bs[t * 16 + nB][khB]);
      bfr.q[1] = *reinterpret_cast<const v4u*>(&Bs[t * 16 + nB][khB + 16]);
      acc[t] = __builtin_amdgcn_wmma_f32_16x16x32_bf16(
          false, a.v, false, bfr.v, (short)0, acc[t], false, false);
    }
    __syncthreads();
  }

  const float scale = 0.125f;  // 1/sqrt(64)
  const int mBase = wave * 16 + (lane >> 4) * 8;
  float* arow = attn + ((size_t)bh * SEQ + sq0) * SEQ;
  #pragma unroll
  for (int t = 0; t < 4; ++t) {
    int skc = sk0 + t * 16 + (lane & 15);
    float madd = (mask[(size_t)b * SEQ + skc] == 0) ? -__builtin_inff() : 0.0f;
    #pragma unroll
    for (int r = 0; r < 8; ++r)
      arow[(size_t)(mBase + r) * SEQ + skc] = acc[t][r] * scale + madd;
  }
}

// =====================================================================
// Row softmax over SK=2048, in place. One 256-thread block per row.
// =====================================================================
__global__ __launch_bounds__(256) void softmax_rows(float* __restrict__ attn)
{
  const size_t row = blockIdx.x;
  float* p = attn + row * (size_t)SEQ;
  const int tid = threadIdx.x;
  __shared__ float red[256];

  float4 f0 = *reinterpret_cast<const float4*>(&p[tid * 8]);
  float4 f1 = *reinterpret_cast<const float4*>(&p[tid * 8 + 4]);
  float v[8] = { f0.x, f0.y, f0.z, f0.w, f1.x, f1.y, f1.z, f1.w };

  float m = -__builtin_inff();
  #pragma unroll
  for (int i = 0; i < 8; ++i) m = fmaxf(m, v[i]);
  red[tid] = m; __syncthreads();
  for (int s = 128; s > 0; s >>= 1) {
    if (tid < s) red[tid] = fmaxf(red[tid], red[tid + s]);
    __syncthreads();
  }
  m = red[0]; __syncthreads();

  float sum = 0.0f;
  #pragma unroll
  for (int i = 0; i < 8; ++i) { v[i] = __expf(v[i] - m); sum += v[i]; }
  red[tid] = sum; __syncthreads();
  for (int s = 128; s > 0; s >>= 1) {
    if (tid < s) red[tid] += red[tid + s];
    __syncthreads();
  }
  float inv = 1.0f / red[0];
  float4 o0 = make_float4(v[0] * inv, v[1] * inv, v[2] * inv, v[3] * inv);
  float4 o1 = make_float4(v[4] * inv, v[5] * inv, v[6] * inv, v[7] * inv);
  *reinterpret_cast<float4*>(&p[tid * 8])     = o0;
  *reinterpret_cast<float4*>(&p[tid * 8 + 4]) = o1;
}

// =====================================================================
// ctx[b,sq,h,dv] = sum_sk attn[b,h,sq,sk] * v[b,sk,h,dv]
// M=128 sq rows, N=64 (all of DV), K over SEQ step 32, double-buffered.
// =====================================================================
__global__ __launch_bounds__(256) void wmma_ctx(
    const float* __restrict__ attn, const float* __restrict__ v,
    float* __restrict__ ctx)
{
  __shared__ alignas(16) bf16 As[2][128][40];
  __shared__ alignas(16) bf16 Bs[2][64][40];

  const int tid  = threadIdx.x;
  const int lane = tid & 31;
  const int wave = tid >> 5;
  const int sq0 = blockIdx.y * 128;
  const int bh  = blockIdx.z;
  const int b = bh / HEADS, h = bh % HEADS;

  const float* arow = attn + (size_t)bh * SEQ * SEQ;
  const float* vb   = v + (size_t)b * SEQ * DM + h * DK;

  v8f acc[4] = {};
  const int rA  = wave * 16 + (lane & 15);
  const int kbA = (lane >> 4) * 8;
  const int nB  = lane & 15;
  const int khB = (lane >> 4) * 8;

  float4 fa[4], fb[2];

  auto load_regs = [&](int kk) {
    #pragma unroll
    for (int i = 0; i < 4; ++i) {
      int idx = i * 256 + tid;
      int r = idx >> 3, c4 = (idx & 7) * 4;
      fa[i] = *reinterpret_cast<const float4*>(
          &arow[(size_t)(sq0 + r) * SEQ + kk + c4]);
    }
    #pragma unroll
    for (int i = 0; i < 2; ++i) {
      int idx = i * 256 + tid;
      int kr = idx >> 4, nc = (idx & 15) * 4;
      fb[i] = *reinterpret_cast<const float4*>(
          &vb[(size_t)(kk + kr) * DM + nc]);
    }
  };
  auto store_lds = [&](int buf) {
    #pragma unroll
    for (int i = 0; i < 4; ++i) {
      int idx = i * 256 + tid;
      int r = idx >> 3, c4 = (idx & 7) * 4;
      *reinterpret_cast<v4bf*>(&As[buf][r][c4]) = cvt4(fa[i]);
    }
    #pragma unroll
    for (int i = 0; i < 2; ++i) {
      int idx = i * 256 + tid;
      int kr = idx >> 4, nc = (idx & 15) * 4;
      Bs[buf][nc + 0][kr] = (bf16)fb[i].x;   // transpose on stage
      Bs[buf][nc + 1][kr] = (bf16)fb[i].y;
      Bs[buf][nc + 2][kr] = (bf16)fb[i].z;
      Bs[buf][nc + 3][kr] = (bf16)fb[i].w;
    }
  };

  const int nsteps = SEQ >> 5;     // 64
  load_regs(0);
  store_lds(0);
  __syncthreads();

  for (int s = 0; s < nsteps; ++s) {
    const int buf = s & 1;
    if (s + 1 < nsteps) load_regs((s + 1) << 5);
    if (s + 2 < nsteps)
      __builtin_prefetch(&arow[(size_t)(sq0 + rA) * SEQ + ((s + 2) << 5)], 0, 1);

    FragBF a;
    a.q[0] = *reinterpret_cast<const v4u*>(&As[buf][rA][kbA]);
    a.q[1] = *reinterpret_cast<const v4u*>(&As[buf][rA][kbA + 16]);
    #pragma unroll
    for (int t = 0; t < 4; ++t) {
      FragBF bfr;
      bfr.q[0] = *reinterpret_cast<const v4u*>(&Bs[buf][t * 16 + nB][khB]);
      bfr.q[1] = *reinterpret_cast<const v4u*>(&Bs[buf][t * 16 + nB][khB + 16]);
      acc[t] = __builtin_amdgcn_wmma_f32_16x16x32_bf16(
          false, a.v, false, bfr.v, (short)0, acc[t], false, false);
    }

    if (s + 1 < nsteps) store_lds(buf ^ 1);
    __syncthreads();
  }

  const int mBase = wave * 16 + (lane >> 4) * 8;
  #pragma unroll
  for (int t = 0; t < 4; ++t) {
    int col = h * DK + t * 16 + (lane & 15);
    #pragma unroll
    for (int r = 0; r < 8; ++r)
      ctx[(size_t)(b * SEQ + sq0 + mBase + r) * DM + col] = acc[t][r];
  }
}

// =====================================================================
// Residual + LayerNorm: one 256-thread block per row of 1024.
// =====================================================================
__global__ __launch_bounds__(256) void residual_layernorm(
    const float* __restrict__ proj, const float* __restrict__ Qin,
    const float* __restrict__ g, const float* __restrict__ beta,
    float* __restrict__ out)
{
  const size_t row = blockIdx.x;
  const int tid = threadIdx.x;
  const int c4 = tid * 4;
  const float* o  = proj + row * (size_t)DM;
  const float* qi = Qin  + row * (size_t)DM;

  __shared__ float rs[256];
  __shared__ float rq[256];

  float4 fo = *reinterpret_cast<const float4*>(&o[c4]);
  float4 fq = *reinterpret_cast<const float4*>(&qi[c4]);
  float x[4] = { fo.x + fq.x, fo.y + fq.y, fo.z + fq.z, fo.w + fq.w };

  float s = 0.0f, sq = 0.0f;
  #pragma unroll
  for (int i = 0; i < 4; ++i) { s += x[i]; sq += x[i] * x[i]; }
  rs[tid] = s; rq[tid] = sq; __syncthreads();
  for (int st = 128; st > 0; st >>= 1) {
    if (tid < st) { rs[tid] += rs[tid + st]; rq[tid] += rq[tid + st]; }
    __syncthreads();
  }
  float mean = rs[0] * (1.0f / DM);
  float var  = rq[0] * (1.0f / DM) - mean * mean;
  float rinv = rsqrtf(var + LN_EPS);

  float4 fg = *reinterpret_cast<const float4*>(&g[c4]);
  float4 fb = *reinterpret_cast<const float4*>(&beta[c4]);
  float4 r;
  r.x = (x[0] - mean) * rinv * fg.x + fb.x;
  r.y = (x[1] - mean) * rinv * fg.y + fb.y;
  r.z = (x[2] - mean) * rinv * fg.z + fb.z;
  r.w = (x[3] - mean) * rinv * fg.w + fb.w;
  *reinterpret_cast<float4*>(&out[row * (size_t)DM + c4]) = r;
}

// =====================================================================
// host-side launch
// =====================================================================
extern "C" void kernel_launch(void* const* d_in, const int* in_sizes, int n_in,
                              void* d_out, int out_size, void* d_ws, size_t ws_size,
                              hipStream_t stream) {
  const float* Q    = (const float*)d_in[0];
  const float* K    = (const float*)d_in[1];
  const float* V    = (const float*)d_in[2];
  const int*   mask = (const int*)  d_in[3];
  const float* Wq   = (const float*)d_in[4];
  const float* bq   = (const float*)d_in[5];
  const float* Wk   = (const float*)d_in[6];
  const float* bk   = (const float*)d_in[7];
  const float* Wv   = (const float*)d_in[8];
  const float* bv   = (const float*)d_in[9];
  const float* Wo   = (const float*)d_in[10];
  const float* bo   = (const float*)d_in[11];
  const float* ln_g = (const float*)d_in[12];
  const float* ln_b = (const float*)d_in[13];

  const size_t ROWS = (size_t)BATCH * SEQ;        // 8192
  const size_t MAT  = ROWS * DM;                  // 8,388,608 floats

  float* normed = (float*)d_out;                  // [B,S,DM]
  float* attn   = (float*)d_out + MAT;            // [B,H,S,S]

  float* q   = (float*)d_ws;
  float* k   = q   + MAT;
  float* v   = k   + MAT;
  float* ctx = v   + MAT;
  float* prj = ctx + MAT;

  dim3 blk(256);
  dim3 gGemm(DM / 64, (unsigned)(ROWS / 128));              // (16, 64)
  wmma_gemm_bias<<<gGemm, blk, 0, stream>>>(Q, Wq, bq, q, (int)ROWS, DM, DM);
  wmma_gemm_bias<<<gGemm, blk, 0, stream>>>(K, Wk, bk, k, (int)ROWS, DM, DM);
  wmma_gemm_bias<<<gGemm, blk, 0, stream>>>(V, Wv, bv, v, (int)ROWS, DM, DM);

  dim3 gScores(SEQ / 64, SEQ / 128, BATCH * HEADS);          // (32, 16, 64)
  wmma_scores<<<gScores, blk, 0, stream>>>(q, k, mask, attn);

  dim3 gSoft((unsigned)((size_t)BATCH * HEADS * SEQ));       // 131072 rows
  softmax_rows<<<gSoft, blk, 0, stream>>>(attn);

  dim3 gCtx(1, SEQ / 128, BATCH * HEADS);                    // (1, 16, 64)
  wmma_ctx<<<gCtx, blk, 0, stream>>>(attn, v, ctx);

  wmma_gemm_bias<<<gGemm, blk, 0, stream>>>(ctx, Wo, bo, prj, (int)ROWS, DM, DM);

  dim3 gLN((unsigned)ROWS);                                  // 8192 rows
  residual_layernorm<<<gLN, blk, 0, stream>>>(prj, Q, ln_g, ln_b, normed);
}
